// SGFormerAttention_86681029968108
// MI455X (gfx1250) — matrix operand order repaired and checked by reading, not compiled, not run on hardware
//
#include <hip/hip_runtime.h>
#include <hip/hip_bf16.h>

typedef __attribute__((ext_vector_type(16))) __bf16 v16bf;
typedef __attribute__((ext_vector_type(8)))  __bf16 v8bf;
typedef __attribute__((ext_vector_type(8)))  float  v8f;
typedef __attribute__((ext_vector_type(4)))  float  f32x4;

#define N_NODES 200000
#define CHN 256
#define HD  64
#define ROWS_PER_BLOCK 128
#define NBLOCKS ((N_NODES + ROWS_PER_BLOCK - 1) / ROWS_PER_BLOCK)   // 1563
#define KSTRIDE 136   // 128 + 8 bf16 pad to break LDS bank alignment
#define NF 200000.0f

// ---- workspace layout (bytes) ----
#define OFF_WQB  ((size_t)0)                    // 64*256 bf16 = 32768
#define OFF_WKB  ((size_t)32768)
#define OFF_WVB  ((size_t)65536)
#define OFF_KVG  ((size_t)98304)                // 64*64 f32 = 16384
#define OFF_KSUM ((size_t)114688)               // 64 f32
#define OFF_SQ   ((size_t)114944)               // 2 f32
#define OFF_KVT  ((size_t)115200)               // 64*64 bf16 = 8192
#define OFF_KST  ((size_t)123392)               // 64 bf16
#define OFF_QB   ((size_t)131072)               // N*64 bf16 = 25,600,000
#define OFF_VWS  ((size_t)(131072 + 25600000))  // N*64 f32  = 51,200,000

__device__ __forceinline__ v8f wmma_bf16(v16bf a, v16bf b, v8f c) {
  return __builtin_amdgcn_wmma_f32_16x16x32_bf16(false, a, false, b, (short)0, c,
                                                 false, false);
}

__device__ __forceinline__ v16bf cat8(v8bf lo, v8bf hi) {
  v16bf r;
#pragma unroll
  for (int i = 0; i < 8; ++i) { r[i] = lo[i]; r[i + 8] = hi[i]; }
  return r;
}

// 8 contiguous f32 -> 8 bf16, non-temporal (x is single-use streaming data)
__device__ __forceinline__ v8bf cvt8nt(const float* p) {
  f32x4 f0 = __builtin_nontemporal_load((const f32x4*)p);
  f32x4 f1 = __builtin_nontemporal_load(((const f32x4*)p) + 1);
  v8bf r;
  r[0] = (__bf16)f0.x; r[1] = (__bf16)f0.y; r[2] = (__bf16)f0.z; r[3] = (__bf16)f0.w;
  r[4] = (__bf16)f1.x; r[5] = (__bf16)f1.y; r[6] = (__bf16)f1.z; r[7] = (__bf16)f1.w;
  return r;
}

__device__ __forceinline__ v8bf ldnt8(const __bf16* p) {
  return __builtin_nontemporal_load((const v8bf*)p);
}

// -------- kernel 0: zero accumulators, convert weights to bf16 --------
__global__ void k_init(const float* __restrict__ Wq, const float* __restrict__ Wk,
                       const float* __restrict__ Wv,
                       __bf16* __restrict__ Wqb, __bf16* __restrict__ Wkb,
                       __bf16* __restrict__ Wvb,
                       float* __restrict__ KVg, float* __restrict__ ksumg,
                       float* __restrict__ sqg) {
  const int i = blockIdx.x * blockDim.x + threadIdx.x;
  if (i < HD * CHN) {
    Wqb[i] = (__bf16)Wq[i];
    Wkb[i] = (__bf16)Wk[i];
    Wvb[i] = (__bf16)Wv[i];
  }
  if (i < HD * HD) KVg[i] = 0.0f;
  if (i < HD) ksumg[i] = 0.0f;
  if (i < 2) sqg[i] = 0.0f;
}

// -------- kernel 1: Q/K/V projections + all global reductions --------
// 128 threads = 4 waves; 32 rows per wave so every weight B-tile load feeds
// two WMMAs (halves the weight-tile cache traffic vs 16 rows/wave).
__global__ __launch_bounds__(128) void k_qkv(
    const float* __restrict__ x,
    const __bf16* __restrict__ Wqb, const __bf16* __restrict__ Wkb,
    const __bf16* __restrict__ Wvb,
    __bf16* __restrict__ Qb, float* __restrict__ Vws,
    float* __restrict__ KVg, float* __restrict__ ksumg, float* __restrict__ sqg) {
  __shared__ __align__(16) __bf16 ldsKT[64 * KSTRIDE];  // K^T tile (64 ch x 128 rows)
  __shared__ __align__(16) __bf16 ldsVT[64 * KSTRIDE];  // V^T tile
  __shared__ float lds_ksum[64];

  const int tid = threadIdx.x;
  const int wave = tid >> 5, lane = tid & 31;
  const int mrow = lane & 15, hi = lane >> 4;
  const int rowbase = blockIdx.x * ROWS_PER_BLOCK + wave * 32;
  const bool active = rowbase < N_NODES;   // N % 32 == 0: all-or-nothing per wave

  if (tid < 64) lds_ksum[tid] = 0.0f;
  __syncthreads();

  v8f aq[2][4], ak[2][4], av[2][4];
#pragma unroll
  for (int rt = 0; rt < 2; ++rt)
#pragma unroll
    for (int nt = 0; nt < 4; ++nt)
#pragma unroll
      for (int v = 0; v < 8; ++v) {
        aq[rt][nt][v] = 0.f; ak[rt][nt][v] = 0.f; av[rt][nt][v] = 0.f;
      }

  if (active) {
    const float* xr0 = x + (size_t)(rowbase + mrow) * CHN;
    const float* xr1 = x + (size_t)(rowbase + 16 + mrow) * CHN;
#pragma unroll
    for (int s = 0; s < 8; ++s) {
      const int kb = s * 32;
      // Two A tiles (rows rowbase..+15 and +16..+31), k contiguous per lane half (16*hi + s), same order as B
      v16bf a0 = cat8(cvt8nt(xr0 + kb + hi * 16), cvt8nt(xr0 + kb + hi * 16 + 8));
      v16bf a1 = cat8(cvt8nt(xr1 + kb + hi * 16), cvt8nt(xr1 + kb + hi * 16 + 8));
#pragma unroll
      for (int nt = 0; nt < 4; ++nt) {
        const int r = nt * 16 + mrow;       // B lane n reads W row n across k
        const int kk = kb + hi * 16;
        v16bf b = *(const v16bf*)(Wqb + r * CHN + kk);
        aq[0][nt] = wmma_bf16(a0, b, aq[0][nt]);
        aq[1][nt] = wmma_bf16(a1, b, aq[1][nt]);
        b = *(const v16bf*)(Wkb + r * CHN + kk);
        ak[0][nt] = wmma_bf16(a0, b, ak[0][nt]);
        ak[1][nt] = wmma_bf16(a1, b, ak[1][nt]);
        b = *(const v16bf*)(Wvb + r * CHN + kk);
        av[0][nt] = wmma_bf16(a0, b, av[0][nt]);
        av[1][nt] = wmma_bf16(a1, b, av[1][nt]);
      }
    }
    // persist Q (bf16 row-major) and V (f32 per-lane tile layout); reduce sums
    float sq = 0.f, sk = 0.f;
#pragma unroll
    for (int rt = 0; rt < 2; ++rt) {
      const int rb = rowbase + rt * 16;
      float* vt = Vws + (size_t)(rb >> 4) * 1024;
#pragma unroll
      for (int nt = 0; nt < 4; ++nt) {
#pragma unroll
        for (int v = 0; v < 8; ++v) {
          const int row = rb + v + 8 * hi;
          Qb[(size_t)row * HD + nt * 16 + mrow] = (__bf16)aq[rt][nt][v];
          vt[(nt * 8 + v) * 32 + lane] = av[rt][nt][v];
          sq += aq[rt][nt][v] * aq[rt][nt][v];
          sk += ak[rt][nt][v] * ak[rt][nt][v];
        }
      }
    }
#pragma unroll
    for (int nt = 0; nt < 4; ++nt) {
      float kcol = 0.f;
#pragma unroll
      for (int rt = 0; rt < 2; ++rt)
#pragma unroll
        for (int v = 0; v < 8; ++v) kcol += ak[rt][nt][v];
      kcol += __shfl_xor(kcol, 16);  // fold lane L with L+16 (same column)
      if (lane < 16) atomicAdd(&lds_ksum[nt * 16 + lane], kcol);
    }
#pragma unroll
    for (int off = 16; off > 0; off >>= 1) {
      sq += __shfl_xor(sq, off);
      sk += __shfl_xor(sk, off);
    }
    if (lane == 0) { atomicAdd(&sqg[0], sq); atomicAdd(&sqg[1], sk); }
  }

  // stage K^T / V^T (inactive waves contribute zeros so KV WMMA is exact)
#pragma unroll
  for (int rt = 0; rt < 2; ++rt)
#pragma unroll
    for (int nt = 0; nt < 4; ++nt)
#pragma unroll
      for (int v = 0; v < 8; ++v) {
        const int col = nt * 16 + mrow;
        const int lr = wave * 32 + rt * 16 + v + 8 * hi;
        ldsKT[col * KSTRIDE + lr] = (__bf16)ak[rt][nt][v];
        ldsVT[col * KSTRIDE + lr] = (__bf16)av[rt][nt][v];
      }
  __syncthreads();

  if (tid < 64) atomicAdd(&ksumg[tid], lds_ksum[tid]);

  // KV += K^T V over this block's 128 rows: 16 tiles of 16x16, 4 per wave
#pragma unroll
  for (int i = 0; i < 4; ++i) {
    const int t = wave * 4 + i;
    const int mt = t >> 2, nt2 = t & 3;
    v8f kv;
#pragma unroll
    for (int v = 0; v < 8; ++v) kv[v] = 0.f;
#pragma unroll
    for (int s = 0; s < 4; ++s) {
      const int kb = s * 32;
      const __bf16* arow = ldsKT + (mt * 16 + mrow) * KSTRIDE;
      const __bf16* brow = ldsVT + (nt2 * 16 + mrow) * KSTRIDE;
      v16bf a = cat8(*(const v8bf*)(arow + kb + hi * 16),
                     *(const v8bf*)(arow + kb + hi * 16 + 8));
      v16bf b = cat8(*(const v8bf*)(brow + kb + hi * 16),
                     *(const v8bf*)(brow + kb + hi * 16 + 8));
      kv = wmma_bf16(a, b, kv);
    }
#pragma unroll
    for (int v = 0; v < 8; ++v)
      atomicAdd(&KVg[(mt * 16 + v + 8 * hi) * HD + nt2 * 16 + mrow], kv[v]);
  }
}

// -------- kernel 2: finalize scalars, scale + transpose KV to bf16 --------
__global__ void k_finalize(const float* __restrict__ KVg, const float* __restrict__ ksumg,
                           const float* __restrict__ sqg,
                           __bf16* __restrict__ KVT, __bf16* __restrict__ kst) {
  const float s = rsqrtf(sqg[0]) * rsqrtf(sqg[1]);  // 1/(||Q|| * ||K||)
  const int t = threadIdx.x;  // 64 threads
  kst[t] = (__bf16)(ksumg[t] * s);
  for (int j = 0; j < HD; ++j) KVT[t * HD + j] = (__bf16)(KVg[j * HD + t] * s);
}

// -------- kernel 3: out = (Q@KV_s + N*V) / (Q.ksum_s + N) --------
__global__ __launch_bounds__(256) void k_out(
    const __bf16* __restrict__ Qb, const float* __restrict__ Vws,
    const __bf16* __restrict__ KVT, const __bf16* __restrict__ kst,
    float* __restrict__ out) {
  const int tid = threadIdx.x;
  const int wave = tid >> 5, lane = tid & 31;
  const int mrow = lane & 15, hi = lane >> 4;
  const int rowbase = blockIdx.x * ROWS_PER_BLOCK + wave * 16;
  if (rowbase >= N_NODES) return;  // wave-uniform: EXEC stays all-ones for WMMA

  v8f num[4], dot;
#pragma unroll
  for (int nt = 0; nt < 4; ++nt)
#pragma unroll
    for (int v = 0; v < 8; ++v) num[nt][v] = 0.f;
#pragma unroll
  for (int v = 0; v < 8; ++v) dot[v] = 0.f;

  const __bf16* qrow = Qb + (size_t)(rowbase + mrow) * HD;
#pragma unroll
  for (int s = 0; s < 2; ++s) {
    const int kb = s * 32;
    v16bf a = cat8(ldnt8(qrow + kb + hi * 16), ldnt8(qrow + kb + hi * 16 + 8));
#pragma unroll
    for (int nt = 0; nt < 4; ++nt) {
      const __bf16* brow = KVT + (nt * 16 + mrow) * HD + kb + hi * 16;
      v16bf b = cat8(*(const v8bf*)(brow), *(const v8bf*)(brow + 8));
      num[nt] = wmma_bf16(a, b, num[nt]);
    }
    // normalizer dot: B tile whose column 0 is scaled ksum
    v16bf bd;
#pragma unroll
    for (int i = 0; i < 16; ++i) bd[i] = (__bf16)0.0f;
    if (mrow == 0) {
      const __bf16* kp = kst + kb + hi * 16;
      bd = cat8(*(const v8bf*)(kp), *(const v8bf*)(kp + 8));
    }
    dot = wmma_bf16(a, bd, dot);
  }

  const float* vt = Vws + (size_t)(rowbase >> 4) * 1024;
  float denom[8];
#pragma unroll
  for (int v = 0; v < 8; ++v) denom[v] = __shfl(dot[v], hi << 4) + NF;
#pragma unroll
  for (int nt = 0; nt < 4; ++nt)
#pragma unroll
    for (int v = 0; v < 8; ++v) {
      const float vv = __builtin_nontemporal_load(vt + (nt * 8 + v) * 32 + lane);
      const float o = (num[nt][v] + NF * vv) / denom[v];
      __builtin_nontemporal_store(o, out + (size_t)(rowbase + v + 8 * hi) * HD +
                                         nt * 16 + mrow);
    }
}

extern "C" void kernel_launch(void* const* d_in, const int* in_sizes, int n_in,
                              void* d_out, int out_size, void* d_ws, size_t ws_size,
                              hipStream_t stream) {
  (void)in_sizes; (void)n_in; (void)out_size; (void)ws_size;
  const float* x  = (const float*)d_in[0];
  const float* Wq = (const float*)d_in[1];
  const float* Wk = (const float*)d_in[2];
  const float* Wv = (const float*)d_in[3];
  float* out = (float*)d_out;

  char* ws = (char*)d_ws;
  __bf16* Wqb  = (__bf16*)(ws + OFF_WQB);
  __bf16* Wkb  = (__bf16*)(ws + OFF_WKB);
  __bf16* Wvb  = (__bf16*)(ws + OFF_WVB);
  float*  KVg  = (float*)(ws + OFF_KVG);
  float*  ksum = (float*)(ws + OFF_KSUM);
  float*  sqg  = (float*)(ws + OFF_SQ);
  __bf16* KVT  = (__bf16*)(ws + OFF_KVT);
  __bf16* kst  = (__bf16*)(ws + OFF_KST);
  __bf16* Qb   = (__bf16*)(ws + OFF_QB);
  float*  Vws  = (float*)(ws + OFF_VWS);

  k_init<<<64, 256, 0, stream>>>(Wq, Wk, Wv, Wqb, Wkb, Wvb, KVg, ksum, sqg);
  k_qkv<<<NBLOCKS, 128, 0, stream>>>(x, Wqb, Wkb, Wvb, Qb, Vws, KVg, ksum, sqg);
  k_finalize<<<1, 64, 0, stream>>>(KVg, ksum, sqg, KVT, kst);
  k_out<<<NBLOCKS, 256, 0, stream>>>(Qb, Vws, KVT, kst, out);
}
